// GCN_hinge_18348100289005
// MI455X (gfx1250) — matrix operand order repaired and checked by reading, not compiled, not run on hardware
//
#include <hip/hip_runtime.h>

// Problem constants (from reference)
#define NN    8192   // nodes
#define NFEAT 512
#define NHID  256
#define NCLS  2
#define KCAT  (3 * NFEAT)   // 1536: concatenated Cheb K dimension

typedef __attribute__((ext_vector_type(4)))  float   v4f;
typedef __attribute__((ext_vector_type(8)))  float   v8f;
typedef __attribute__((ext_vector_type(16))) __bf16  v16bf;

// ---------------------------------------------------------------------------
// A-tile loader: 16x32 bf16 A tile for v_wmma_f32_16x16x32_bf16, converted
// on the fly from row-major f32. K order used here is contiguous (B matches):
//   lanes 0-15 : M = lane,    K = {0..15}
//   lanes 16-31: M = lane-16, K = {16..31}
// rowptr points at A[row][0]; kbase = k0 + (lane&16 ? 16 : 0).
// ---------------------------------------------------------------------------
__device__ __forceinline__ v16bf load_a_f32_to_bf16(const float* __restrict__ rowptr,
                                                    int kbase) {
  v4f p0 = *(const v4f*)(rowptr + kbase);
  v4f p1 = *(const v4f*)(rowptr + kbase + 4);
  v4f p2 = *(const v4f*)(rowptr + kbase + 8);
  v4f p3 = *(const v4f*)(rowptr + kbase + 12);
  v16bf a;
#pragma unroll
  for (int e = 0; e < 4; ++e) {
    a[e]      = (__bf16)p0[e];
    a[e + 4]  = (__bf16)p1[e];
    a[e + 8]  = (__bf16)p2[e];
    a[e + 12] = (__bf16)p3[e];
  }
  return a;
}

// ---------------------------------------------------------------------------
// K0: deg[i] = sum_j adj[i][j];  dinv[i] = rsqrt(max(deg,1e-12))
// ---------------------------------------------------------------------------
__global__ __launch_bounds__(256) void rowsum_dinv_kernel(const float* __restrict__ adj,
                                                          float* __restrict__ dinv) {
  __shared__ float s[256];
  const int i   = blockIdx.x;
  const int tid = threadIdx.x;
  const float* row = adj + (size_t)i * NN;
  float acc = 0.f;
  for (int j = tid * 4; j < NN; j += 256 * 4) {
    v4f v = *(const v4f*)(row + j);
    acc += v[0] + v[1] + v[2] + v[3];
  }
  s[tid] = acc;
  __syncthreads();
  for (int st = 128; st > 0; st >>= 1) {
    if (tid < st) s[tid] += s[tid + st];
    __syncthreads();
  }
  if (tid == 0) dinv[i] = rsqrtf(fmaxf(s[0], 1e-12f));
}

// ---------------------------------------------------------------------------
// K1: XsT[f][j] = bf16( dinv[j] * x[j][f] )   (transposed, pre-scaled B operand)
// grid: (NN/256, NFEAT)
// ---------------------------------------------------------------------------
__global__ __launch_bounds__(256) void build_xsT_kernel(const float* __restrict__ x,
                                                        const float* __restrict__ dinv,
                                                        __bf16* __restrict__ xsT) {
  const int j = blockIdx.x * 256 + threadIdx.x;
  const int f = blockIdx.y;
  xsT[(size_t)f * NN + j] = (__bf16)(dinv[j] * x[(size_t)j * NFEAT + f]);
}

// ---------------------------------------------------------------------------
// K1b: WcatT[h][k] = bf16( cheb_W_flat[k*NHID + h] ), k in [0,1536)
// cheb_W flat layout [3][512][256] == [k_global][h] already.
// ---------------------------------------------------------------------------
__global__ __launch_bounds__(256) void build_wcatT_kernel(const float* __restrict__ chebW,
                                                          __bf16* __restrict__ wcatT) {
  const int idx = blockIdx.x * 256 + threadIdx.x;   // idx = h*1536 + k
  const int h = idx / KCAT;
  const int k = idx % KCAT;
  wcatT[idx] = (__bf16)(chebW[(size_t)k * NHID + h]);
}

// ---------------------------------------------------------------------------
// K2/K3: Y = adj @ Bs  (bf16 WMMA, f32 accumulate), fused normalization epilogue.
//   mode 1: Z1 = -dinv_i * Y ;  Z1sT[f][i] = bf16(dinv_i * Z1[i][f])
//   mode 2: Z2 = -2*dinv_i*Y - x[i][f]
// Block = 256 thr = 8 waves; block covers 16 rows x 512 cols (adj streamed once);
// wave covers 16 x 64 (4 accumulator tiles).
// ---------------------------------------------------------------------------
__global__ __launch_bounds__(256) void gemm_adj_kernel(const float* __restrict__ adj,
                                                       const __bf16* __restrict__ BT,
                                                       const float* __restrict__ dinv,
                                                       const float* __restrict__ x,
                                                       float* __restrict__ Zout,
                                                       __bf16* __restrict__ ZsT,
                                                       int mode) {
  const int lane  = threadIdx.x & 31;
  const int wave  = threadIdx.x >> 5;
  const int ibase = blockIdx.x * 16;
  const int nbase = wave * 64;
  const int l15   = lane & 15;
  const int akoff = (lane & 16) ? 16 : 0;   // A-tile K sub-offset (contiguous)
  const int bkoff = (lane & 16) ? 16 : 0;   // B-tile K sub-offset
  const float* arow = adj + (size_t)(ibase + l15) * NN;

  v8f acc[4] = {v8f{}, v8f{}, v8f{}, v8f{}};

  for (int k0 = 0; k0 < NN; k0 += 32) {
    __builtin_prefetch(arow + k0 + 512, 0, 0);       // global_prefetch_b8
    v16bf a = load_a_f32_to_bf16(arow, k0 + akoff);
#pragma unroll
    for (int t = 0; t < 4; ++t) {
      const __bf16* bp = BT + (size_t)(nbase + t * 16 + l15) * NN + k0 + bkoff;
      v16bf b = *(const v16bf*)bp;                   // K-contiguous (2x b128)
      acc[t] = __builtin_amdgcn_wmma_f32_16x16x32_bf16(
          false, a, false, b, (short)0, acc[t], false, false);
    }
  }

  // C/D layout: lanes 0-15 -> N=lane, M=r ; lanes 16-31 -> N=lane-16, M=8+r
  const int rlo = (lane & 16) ? 8 : 0;
#pragma unroll
  for (int t = 0; t < 4; ++t) {
    const int col = nbase + t * 16 + l15;
#pragma unroll
    for (int r = 0; r < 8; ++r) {
      const int grow = ibase + rlo + r;
      const float di = dinv[grow];
      const float y  = acc[t][r];
      if (mode == 1) {
        const float z1 = -di * y;
        Zout[(size_t)grow * NFEAT + col] = z1;
        ZsT[(size_t)col * NN + grow]     = (__bf16)(di * z1);
      } else {
        Zout[(size_t)grow * NFEAT + col] =
            -2.0f * di * y - x[(size_t)grow * NFEAT + col];
      }
    }
  }
}

// ---------------------------------------------------------------------------
// K4: h = relu( [Z0|Z1|Z2] @ Wcat + cheb_b )   (M=8192, K=1536, N=256)
// Wave covers 16 x 64; 4 col-groups x 512 row-strips = 2048 waves = 256 blocks.
// ---------------------------------------------------------------------------
__global__ __launch_bounds__(256) void gemm_h_kernel(const float* __restrict__ x,
                                                     const float* __restrict__ Z1,
                                                     const float* __restrict__ Z2,
                                                     const __bf16* __restrict__ WT,
                                                     const float* __restrict__ chebB,
                                                     float* __restrict__ H) {
  const int lane  = threadIdx.x & 31;
  const int gwave = blockIdx.x * 8 + (threadIdx.x >> 5);
  const int ibase = (gwave >> 2) * 16;   // row strip
  const int nbase = (gwave & 3) * 64;    // col group
  const int l15   = lane & 15;
  const int akoff = (lane & 16) ? 16 : 0;
  const int bkoff = (lane & 16) ? 16 : 0;
  const int arow  = ibase + l15;
  const float* seg0 = x  + (size_t)arow * NFEAT;
  const float* seg1 = Z1 + (size_t)arow * NFEAT;
  const float* seg2 = Z2 + (size_t)arow * NFEAT;

  v8f acc[4] = {v8f{}, v8f{}, v8f{}, v8f{}};

  for (int k0 = 0; k0 < KCAT; k0 += 32) {
    const int s  = k0 >> 9;          // segment: 0->Z0(x), 1->Z1, 2->Z2
    const int kk = k0 & (NFEAT - 1);
    const float* seg = (s == 0) ? seg0 : ((s == 1) ? seg1 : seg2);
    v16bf a = load_a_f32_to_bf16(seg, kk + akoff);
#pragma unroll
    for (int t = 0; t < 4; ++t) {
      const __bf16* bp = WT + (size_t)(nbase + t * 16 + l15) * KCAT + k0 + bkoff;
      v16bf b = *(const v16bf*)bp;
      acc[t] = __builtin_amdgcn_wmma_f32_16x16x32_bf16(
          false, a, false, b, (short)0, acc[t], false, false);
    }
  }

  const int rlo = (lane & 16) ? 8 : 0;
#pragma unroll
  for (int t = 0; t < 4; ++t) {
    const int col = nbase + t * 16 + l15;
    const float bias = chebB[col];
#pragma unroll
    for (int r = 0; r < 8; ++r) {
      const int grow = ibase + rlo + r;
      H[(size_t)grow * NHID + col] = fmaxf(acc[t][r] + bias, 0.0f);
    }
  }
}

// ---------------------------------------------------------------------------
// K5: T = H @ gc2_W   (8192x256 @ 256x2, no bias here)
// ---------------------------------------------------------------------------
__global__ __launch_bounds__(256) void hw2_kernel(const float* __restrict__ H,
                                                  const float* __restrict__ W2,
                                                  float* __restrict__ T) {
  const int n = blockIdx.x * 256 + threadIdx.x;
  const float* hrow = H + (size_t)n * NHID;
  float a0 = 0.f, a1 = 0.f;
  for (int k = 0; k < NHID; ++k) {
    const float hv = hrow[k];
    a0 += hv * W2[k * 2 + 0];
    a1 += hv * W2[k * 2 + 1];
  }
  T[n * 2 + 0] = a0;
  T[n * 2 + 1] = a1;
}

// ---------------------------------------------------------------------------
// K6: logits[i][c] = sum_j adj[i][j] * T[j][c] + gc2_b[c]   (block per row)
// ---------------------------------------------------------------------------
__global__ __launch_bounds__(256) void logits_kernel(const float* __restrict__ adj,
                                                     const float* __restrict__ T,
                                                     const float* __restrict__ b2,
                                                     float* __restrict__ L) {
  __shared__ float s0[256];
  __shared__ float s1[256];
  const int i   = blockIdx.x;
  const int tid = threadIdx.x;
  const float* row = adj + (size_t)i * NN;
  float a0 = 0.f, a1 = 0.f;
  for (int j = tid; j < NN; j += 256) {
    const float av = row[j];
    a0 += av * T[j * 2 + 0];
    a1 += av * T[j * 2 + 1];
  }
  s0[tid] = a0;
  s1[tid] = a1;
  __syncthreads();
  for (int st = 128; st > 0; st >>= 1) {
    if (tid < st) { s0[tid] += s0[tid + st]; s1[tid] += s1[tid + st]; }
    __syncthreads();
  }
  if (tid == 0) {
    L[i * 2 + 0] = s0[0] + b2[0];
    L[i * 2 + 1] = s1[0] + b2[1];
  }
}

// ---------------------------------------------------------------------------
// K7: out[c] = max_i logits[i][c]
// ---------------------------------------------------------------------------
__global__ __launch_bounds__(256) void max_kernel(const float* __restrict__ L,
                                                  float* __restrict__ out) {
  __shared__ float m0[256];
  __shared__ float m1[256];
  const int tid = threadIdx.x;
  float a0 = -3.4e38f, a1 = -3.4e38f;
  for (int i = tid; i < NN; i += 256) {
    a0 = fmaxf(a0, L[i * 2 + 0]);
    a1 = fmaxf(a1, L[i * 2 + 1]);
  }
  m0[tid] = a0;
  m1[tid] = a1;
  __syncthreads();
  for (int st = 128; st > 0; st >>= 1) {
    if (tid < st) {
      m0[tid] = fmaxf(m0[tid], m0[tid + st]);
      m1[tid] = fmaxf(m1[tid], m1[tid + st]);
    }
    __syncthreads();
  }
  if (tid == 0) { out[0] = m0[0]; out[1] = m1[0]; }
}

// ---------------------------------------------------------------------------
extern "C" void kernel_launch(void* const* d_in, const int* in_sizes, int n_in,
                              void* d_out, int out_size, void* d_ws, size_t ws_size,
                              hipStream_t stream) {
  const float* x     = (const float*)d_in[0];   // [8192,512]
  const float* adj   = (const float*)d_in[1];   // [8192,8192]
  const float* chebW = (const float*)d_in[2];   // [3,512,256]
  const float* chebB = (const float*)d_in[3];   // [256]
  const float* gc2W  = (const float*)d_in[4];   // [256,2]
  const float* gc2B  = (const float*)d_in[5];   // [2]
  float* out = (float*)d_out;                   // [2]

  // Workspace bump allocation (all 256B-aligned sizes)
  char* ws = (char*)d_ws;
  float*  dinv  = (float*)ws;  ws += (size_t)NN * sizeof(float);
  __bf16* xsT   = (__bf16*)ws; ws += (size_t)NFEAT * NN * sizeof(__bf16);
  __bf16* z1sT  = (__bf16*)ws; ws += (size_t)NFEAT * NN * sizeof(__bf16);
  __bf16* wcatT = (__bf16*)ws; ws += (size_t)NHID * KCAT * sizeof(__bf16);
  float*  Z1    = (float*)ws;  ws += (size_t)NN * NFEAT * sizeof(float);
  float*  Z2    = (float*)ws;  ws += (size_t)NN * NFEAT * sizeof(float);
  float*  H     = (float*)ws;  ws += (size_t)NN * NHID * sizeof(float);
  float*  T     = (float*)ws;  ws += (size_t)NN * NCLS * sizeof(float);
  float*  L     = (float*)ws;  ws += (size_t)NN * NCLS * sizeof(float);

  // K0: degrees -> dinv
  rowsum_dinv_kernel<<<NN, 256, 0, stream>>>(adj, dinv);
  // K1: XsT = bf16( (dinv .* x)^T )
  build_xsT_kernel<<<dim3(NN / 256, NFEAT), 256, 0, stream>>>(x, dinv, xsT);
  // K1b: WcatT bf16
  build_wcatT_kernel<<<(NHID * KCAT) / 256, 256, 0, stream>>>(chebW, wcatT);
  // K2: Z1 = -(A_norm @ x);   also Z1sT = bf16(dinv .* Z1)^T
  gemm_adj_kernel<<<NN / 16, 256, 0, stream>>>(adj, xsT, dinv, x, Z1, z1sT, 1);
  // K3: Z2 = -2*(A_norm @ Z1) - Z0
  gemm_adj_kernel<<<NN / 16, 256, 0, stream>>>(adj, z1sT, dinv, x, Z2, z1sT, 2);
  // K4: h = relu([Z0|Z1|Z2] @ Wcat + b)
  gemm_h_kernel<<<(NN / 16) * (NHID / 64) / 8, 256, 0, stream>>>(x, Z1, Z2, wcatT,
                                                                 chebB, H);
  // K5: T = H @ gc2_W
  hw2_kernel<<<NN / 256, 256, 0, stream>>>(H, gc2W, T);
  // K6: logits = adj @ T + gc2_b
  logits_kernel<<<NN, 256, 0, stream>>>(adj, T, gc2B, L);
  // K7: column max -> out[2]
  max_kernel<<<1, 256, 0, stream>>>(L, out);
}
